// Copynet_7464653160954
// MI455X (gfx1250) — matrix operand order repaired and checked by reading, not compiled, not run on hardware
//
#include <hip/hip_runtime.h>
#include <hip/hip_bf16.h>
#include <stdint.h>
#include <stddef.h>

// ---------------------------------------------------------------------------
// CDNA5 (gfx1250) pointer-generator seq2seq forward pass.
// Every GEMM goes through v_wmma_f32_16x16x32_f16 (f16 operands, f32 acc):
//  - weights converted to f16 once per launch into workspace (out2_W drops to
//    32.8 MB f16 -> resident in the 192 MB L2 across all 64 decoder steps)
//  - wave computes a 16x64 C tile (4 WMMA accumulators) so the M=16 activation
//    fragment is loaded once per K-step and reused 4x -> 1.6x fewer vmem ops
//    per WMMA than a 16x16 tiling.
// ---------------------------------------------------------------------------

typedef _Float16 h16;
typedef __attribute__((ext_vector_type(16))) _Float16 v16h;
typedef __attribute__((ext_vector_type(8)))  float    v8f;

constexpr int kB    = 16;
constexpr int kLE   = 400;    // L_ENC
constexpr int kLD   = 64;     // L_DEC
constexpr int kV    = 32000;
constexpr int kE    = 256;
constexpr int kH    = 512;
constexpr int kH2   = 1024;
constexpr int kOOV  = 30;
constexpr int kVX   = kV + kOOV;   // 32030

__device__ __forceinline__ float sigf(float x) { return 1.0f / (1.0f + __expf(-x)); }

// ---------------------------------------------------------------------------
// WMMA GEMM: C[M,N] = act( A[M,K] * W[N,K]^T + bias[N] )
// A, W are f16 row-major. C is f32; optional f16 mirror Ch (same ldc).
// Requires M%16==0, N%64==0, K%32==0. One wave per 16x64 C tile.
// ---------------------------------------------------------------------------
__global__ void gemm64_k(const h16* __restrict__ A, int lda,
                         const h16* __restrict__ W, int ldw,
                         const float* __restrict__ bias,
                         float* __restrict__ C, h16* __restrict__ Ch,
                         int ldc, int N, int K, int act, int tiles)
{
    int wid = (int)((blockIdx.x * blockDim.x + threadIdx.x) >> 5);
    if (wid >= tiles) return;                    // wave-uniform: EXEC stays all-1s
    int lane = threadIdx.x & 31;
    int ntiles = N >> 6;                         // 64-wide tiles
    int tn = wid % ntiles;
    int tm = wid / ntiles;
    int m0 = tm << 4, n0 = tn << 6;
    int hh  = lane >> 4;                         // half-wave select
    int l15 = lane & 15;

    v8f acc0 = {}, acc1 = {}, acc2 = {}, acc3 = {};
    const h16* arow = A + (size_t)(m0 + l15) * lda;
    const h16* brow0 = W + (size_t)(n0      + l15) * ldw;
    const h16* brow1 = W + (size_t)(n0 + 16 + l15) * ldw;
    const h16* brow2 = W + (size_t)(n0 + 32 + l15) * ldw;
    const h16* brow3 = W + (size_t)(n0 + 48 + l15) * ldw;

    for (int k0 = 0; k0 < K; k0 += 32) {
        v16h a, b0, b1, b2, b3;
#pragma unroll
        for (int v = 0; v < 8; ++v) {
            // A 16x32 f16 layout (ISA 7.12.2): VGPR v<4 -> K = 2v + 8*half, v>=4 -> +16
            int kb = (v < 4) ? (k0 + 2 * v + 8 * hh)
                             : (k0 + 16 + 2 * (v - 4) + 8 * hh);
            a[2 * v]     = arow[kb];
            a[2 * v + 1] = arow[kb + 1];
            // B 32x16: same K mapping as A (VGPR v<4 -> K = 2v + 8*half, v>=4 -> +16)
            int kc = kb;
            b0[2 * v] = brow0[kc]; b0[2 * v + 1] = brow0[kc + 1];
            b1[2 * v] = brow1[kc]; b1[2 * v + 1] = brow1[kc + 1];
            b2[2 * v] = brow2[kc]; b2[2 * v + 1] = brow2[kc + 1];
            b3[2 * v] = brow3[kc]; b3[2 * v + 1] = brow3[kc + 1];
        }
        acc0 = __builtin_amdgcn_wmma_f32_16x16x32_f16(false, a, false, b0, (short)0, acc0, false, false);
        acc1 = __builtin_amdgcn_wmma_f32_16x16x32_f16(false, a, false, b1, (short)0, acc1, false, false);
        acc2 = __builtin_amdgcn_wmma_f32_16x16x32_f16(false, a, false, b2, (short)0, acc2, false, false);
        acc3 = __builtin_amdgcn_wmma_f32_16x16x32_f16(false, a, false, b3, (short)0, acc3, false, false);
    }

#pragma unroll
    for (int q = 0; q < 4; ++q) {
        v8f acc = (q == 0) ? acc0 : (q == 1) ? acc1 : (q == 2) ? acc2 : acc3;
        int n = n0 + 16 * q + l15;
        float bv = bias ? bias[n] : 0.0f;
#pragma unroll
        for (int v = 0; v < 8; ++v) {
            int m = m0 + v + 8 * hh;             // C/D layout: VGPR v -> rows v, v+8
            float val = acc[v] + bv;
            if (act) val = fmaxf(val, 0.0f);
            C[(size_t)m * ldc + n] = val;
            if (Ch) Ch[(size_t)m * ldc + n] = (h16)val;
        }
    }
}

// ---------------------------------------------------------------------------
// Small utility kernels
// ---------------------------------------------------------------------------
__global__ void cvt_f16_k(const float* __restrict__ s, h16* __restrict__ d, long n) {
    long i = (long)blockIdx.x * blockDim.x + threadIdx.x;
    if (i < n) d[i] = (h16)s[i];
}
__global__ void zero_f32_k(float* p, long n) {
    long i = (long)blockIdx.x * blockDim.x + threadIdx.x;
    if (i < n) p[i] = 0.0f;
}
__global__ void zero_f16_k(h16* p, long n) {
    long i = (long)blockIdx.x * blockDim.x + threadIdx.x;
    if (i < n) p[i] = (h16)0.0f;
}
// concat [Wih | Whh] per output row, f32 -> f16
__global__ void catw_k(const float* __restrict__ Wih, const float* __restrict__ Whh,
                       h16* __restrict__ out, int N, int K1, int K2) {
    int K = K1 + K2;
    long n = (long)N * K;
    long i = (long)blockIdx.x * blockDim.x + threadIdx.x;
    if (i >= n) return;
    int row = (int)(i / K), col = (int)(i % K);
    float v = (col < K1) ? Wih[(long)row * K1 + col]
                         : Whh[(long)row * K2 + (col - K1)];
    out[i] = (h16)v;
}
// pack [a | b] rows -> f16 (f32 sources, independent strides)
__global__ void pack2f_k(const float* __restrict__ a, int lda, int ca,
                         const float* __restrict__ bsrc, int ldb, int cb,
                         h16* __restrict__ out, int rows) {
    int cols = ca + cb;
    int i = blockIdx.x * blockDim.x + threadIdx.x;
    if (i >= rows * cols) return;
    int r = i / cols, j = i % cols;
    float v = (j < ca) ? a[(size_t)r * lda + j] : bsrc[(size_t)r * ldb + (j - ca)];
    out[i] = (h16)v;
}
// pack [a | b] rows -> f16 (f16 sources, tight strides)
__global__ void pack2h_k(const h16* __restrict__ a, int ca,
                         const h16* __restrict__ bsrc, int cb,
                         h16* __restrict__ out, int rows) {
    int cols = ca + cb;
    int i = blockIdx.x * blockDim.x + threadIdx.x;
    if (i >= rows * cols) return;
    int r = i / cols, j = i % cols;
    out[i] = (j < ca) ? a[(size_t)r * ca + j] : bsrc[(size_t)r * cb + (j - ca)];
}

// encoder step input: xh[b] = [ emb[tok(b,t)] | h_prev[b] ]  (f16)
__global__ void enc_x_k(const float* __restrict__ emb, const int* __restrict__ toks,
                        int t, int rev, const h16* __restrict__ hprev,
                        h16* __restrict__ xh) {
    int i = blockIdx.x * blockDim.x + threadIdx.x;     // kB * (kE + kH)
    if (i >= kB * (kE + kH)) return;
    int b = i / (kE + kH), j = i % (kE + kH);
    if (j < kE) {
        int pos = rev ? (kLE - 1 - t) : t;
        int tok = toks[b * kLE + pos];
        xh[i] = (h16)emb[(long)tok * kE + j];
    } else {
        xh[i] = hprev[b * kH + (j - kE)];
    }
}
// decoder xc input: [ c_t1 | emb[dec_input(b,t)] ]  (f16, width 1280)
__global__ void dec_x_k(const float* __restrict__ ctx, const float* __restrict__ emb,
                        const int* __restrict__ toks, int t, h16* __restrict__ out) {
    int i = blockIdx.x * blockDim.x + threadIdx.x;     // kB * (kH2 + kE)
    if (i >= kB * (kH2 + kE)) return;
    int b = i / (kH2 + kE), j = i % (kH2 + kE);
    float v;
    if (j < kH2) v = ctx[b * kH2 + j];
    else {
        int tok = toks[b * kLD + t];
        v = emb[(long)tok * kE + (j - kH2)];
    }
    out[i] = (h16)v;
}

// LSTM cell (gate order i,f,g,o). G is [B,4H] pre-activation without bias.
__global__ void cell_k(const float* __restrict__ G, const float* __restrict__ bias,
                       float* __restrict__ cstate, h16* __restrict__ hstate,
                       h16* __restrict__ hout, int hstride,
                       float* __restrict__ shat32, h16* __restrict__ shat16) {
    int i = blockIdx.x * blockDim.x + threadIdx.x;     // kB * kH
    if (i >= kB * kH) return;
    int b = i / kH, j = i % kH;
    const float* g = G + (size_t)b * 4 * kH;
    float gi = g[j] + bias[j];
    float gf = g[kH + j] + bias[kH + j];
    float gg = g[2 * kH + j] + bias[2 * kH + j];
    float go = g[3 * kH + j] + bias[3 * kH + j];
    float c = sigf(gf) * cstate[i] + sigf(gi) * tanhf(gg);
    float h = sigf(go) * tanhf(c);
    cstate[i] = c;
    hstate[i] = (h16)h;
    if (hout) hout[(size_t)b * hstride + j] = (h16)h;
    if (shat32) { shat32[b * kH2 + j] = h; shat32[b * kH2 + kH + j] = c; }
    if (shat16) { shat16[b * kH2 + j] = (h16)h; shat16[b * kH2 + kH + j] = (h16)c; }
}

// scores[b,l] = v . tanh(enc_fea[b,l,:] + dec_fea[b,:] + cov[b,l]*W_c)  (wave per (b,l))
__global__ void attn_score_k(const float* __restrict__ encfea, const float* __restrict__ decfea,
                             const float* __restrict__ cov, const float* __restrict__ Wc,
                             const float* __restrict__ vvec, float* __restrict__ scores) {
    int wid = (int)((blockIdx.x * blockDim.x + threadIdx.x) >> 5);
    if (wid >= kB * kLE) return;
    int lane = threadIdx.x & 31;
    int b = wid / kLE;
    const float* ef = encfea + (size_t)wid * kH2;
    const float* df = decfea + (size_t)b * kH2;
    float cv = cov[wid];
    float s = 0.0f;
    for (int j = lane; j < kH2; j += 32)
        s += tanhf(ef[j] + df[j] + cv * Wc[j]) * vvec[j];
    for (int o = 16; o; o >>= 1) s += __shfl_down(s, o, 32);
    if (lane == 0) scores[wid] = s;
}

// softmax over L, mask after softmax (as in reference); coverage bookkeeping.
__global__ void attn_softmax_k(const float* __restrict__ scores, const float* __restrict__ mask,
                               float* __restrict__ attn, float* __restrict__ cov,
                               float* __restrict__ covloss, float* __restrict__ covterm,
                               int updateLoss) {
    __shared__ float red[512];
    int b = blockIdx.x, tid = threadIdx.x;
    float sc = (tid < kLE) ? scores[b * kLE + tid] : -1e30f;
    red[tid] = sc; __syncthreads();
    for (int s = 256; s; s >>= 1) { if (tid < s) red[tid] = fmaxf(red[tid], red[tid + s]); __syncthreads(); }
    float mx = red[0]; __syncthreads();
    float ex = (tid < kLE) ? __expf(sc - mx) : 0.0f;
    red[tid] = ex; __syncthreads();
    for (int s = 256; s; s >>= 1) { if (tid < s) red[tid] += red[tid + s]; __syncthreads(); }
    float inv = 1.0f / red[0]; __syncthreads();
    float a = 0.0f, mn = 0.0f;
    if (tid < kLE) {
        a = ex * inv * mask[b * kLE + tid];
        attn[b * kLE + tid] = a;
        if (updateLoss) mn = fminf(a, covloss[b * kLE + tid]);   // old covloss
        float nc = cov[b * kLE + tid] + a;
        cov[b * kLE + tid] = nc;
        if (updateLoss) covloss[b * kLE + tid] = nc;
    }
    if (updateLoss) {
        red[tid] = mn; __syncthreads();
        for (int s = 256; s; s >>= 1) { if (tid < s) red[tid] += red[tid + s]; __syncthreads(); }
        if (tid == 0) covterm[b] = red[0];
    }
}

// c_t[b,h] = sum_l attn[b,l] * z_enc[b,l,h]   (block per b)
__global__ void context_k(const float* __restrict__ attn, const float* __restrict__ zenc,
                          float* __restrict__ ctx) {
    __shared__ float al[kLE];
    int b = blockIdx.x, tid = threadIdx.x;                 // 256 threads
    for (int l = tid; l < kLE; l += 256) al[l] = attn[b * kLE + l];
    __syncthreads();
    float acc[4] = {0.f, 0.f, 0.f, 0.f};
    for (int l = 0; l < kLE; ++l) {
        const float* z = zenc + ((size_t)b * kLE + l) * kH2;
        float a = al[l];
#pragma unroll
        for (int q = 0; q < 4; ++q) acc[q] += a * z[tid + 256 * q];
    }
#pragma unroll
    for (int q = 0; q < 4; ++q) ctx[b * kH2 + tid + 256 * q] = acc[q];
}

// p_gen[b] = sigmoid( [c_t | s_hat | x] . pg_W + pg_b )
__global__ void pgen_k(const float* __restrict__ ctx, const float* __restrict__ shat,
                       const float* __restrict__ x, const float* __restrict__ pgW,
                       const float* __restrict__ pgb, float* __restrict__ pgen) {
    __shared__ float red[256];
    int b = blockIdx.x, tid = threadIdx.x;
    float s = 0.0f;
    for (int j = tid; j < kH2; j += 256) s += ctx[b * kH2 + j] * pgW[j];
    for (int j = tid; j < kH2; j += 256) s += shat[b * kH2 + j] * pgW[kH2 + j];
    for (int j = tid; j < kE; j += 256)  s += x[b * kE + j] * pgW[2 * kH2 + j];
    red[tid] = s; __syncthreads();
    for (int st = 128; st; st >>= 1) { if (tid < st) red[tid] += red[tid + st]; __syncthreads(); }
    if (tid == 0) pgen[b] = sigf(red[0] + pgb[0]);
}

// final[b, :V] = p_gen * softmax(logits); zero OOV tail.  (block per b)
__global__ void vocab_dist_k(const float* __restrict__ logits, const float* __restrict__ pgen,
                             float* __restrict__ fin) {
    __shared__ float red[512];
    int b = blockIdx.x, tid = threadIdx.x;
    const float* lg = logits + (size_t)b * kV;
    float mx = -1e30f;
    for (int v = tid; v < kV; v += 512) mx = fmaxf(mx, lg[v]);
    red[tid] = mx; __syncthreads();
    for (int s = 256; s; s >>= 1) { if (tid < s) red[tid] = fmaxf(red[tid], red[tid + s]); __syncthreads(); }
    mx = red[0]; __syncthreads();
    float sum = 0.0f;
    for (int v = tid; v < kV; v += 512) sum += __expf(lg[v] - mx);
    red[tid] = sum; __syncthreads();
    for (int s = 256; s; s >>= 1) { if (tid < s) red[tid] += red[tid + s]; __syncthreads(); }
    float scale = pgen[b] / red[0];
    float* fb = fin + (size_t)b * kVX;
    for (int v = tid; v < kV; v += 512) fb[v] = __expf(lg[v] - mx) * scale;
    for (int v = kV + tid; v < kVX; v += 512) fb[v] = 0.0f;
}

// copy mechanism: final[b, ext[b,l]] += (1-p_gen[b]) * attn[b,l]
__global__ void scatter_k(const float* __restrict__ attn, const float* __restrict__ pgen,
                          const int* __restrict__ ext, float* __restrict__ fin) {
    int i = blockIdx.x * blockDim.x + threadIdx.x;
    if (i >= kB * kLE) return;
    int b = i / kLE;
    atomicAdd(&fin[(size_t)b * kVX + ext[i]], (1.0f - pgen[b]) * attn[i]);
}

__global__ void nll_k(const float* __restrict__ fin, const int* __restrict__ tgt,
                      const float* __restrict__ mask, const float* __restrict__ covterm,
                      int t, float* __restrict__ acc) {
    int b = threadIdx.x;
    if (b >= kB) return;
    float p = fin[(size_t)b * kVX + tgt[b * kLD + t]];
    float snll = -logf(p + 1e-12f) + covterm[b];
    acc[b] += mask[b * kLD + t] * snll;
}

__global__ void finalize_k(const float* __restrict__ acc, const float* __restrict__ mask,
                           float* __restrict__ out) {
    __shared__ float red[32];
    int b = threadIdx.x;
    float v = 0.0f;
    if (b < kB) {
        float ms = 0.0f;
        for (int t = 0; t < kLD; ++t) ms += mask[b * kLD + t];
        v = acc[b] / ms;
    }
    red[b] = v; __syncthreads();
    if (b == 0) {
        float s = 0.0f;
        for (int i = 0; i < kB; ++i) s += red[i];
        out[0] = s / (float)kB;     // nll
        out[1] = 0.0f;              // kld
    }
}

// ---------------------------------------------------------------------------
// Host: deterministic workspace arena + launch sequence
// ---------------------------------------------------------------------------
namespace {
struct Arena {
    char* base; size_t off;
    void* get(size_t bytes) {
        size_t o = (off + 255) & ~(size_t)255;
        off = o + bytes;
        return base + o;
    }
};
inline int blks(long n, int bs = 256) { return (int)((n + bs - 1) / bs); }
} // namespace

extern "C" void kernel_launch(void* const* d_in, const int* in_sizes, int n_in,
                              void* d_out, int out_size, void* d_ws, size_t ws_size,
                              hipStream_t stream) {
    (void)in_sizes; (void)n_in; (void)out_size; (void)ws_size;

    // ---- inputs (setup_inputs() dict order; params leaves in insertion order)
    const float* enc_mask  = (const float*)d_in[0];
    const float* dec_mask  = (const float*)d_in[1];
    const float* emb       = (const float*)d_in[2];
    const float* enc_Wih_f = (const float*)d_in[3];
    const float* enc_Whh_f = (const float*)d_in[4];
    const float* enc_b_f   = (const float*)d_in[5];
    const float* enc_Wih_b = (const float*)d_in[6];
    const float* enc_Whh_b = (const float*)d_in[7];
    const float* enc_b_b   = (const float*)d_in[8];
    const float* h2l_W     = (const float*)d_in[9];
    const float* h2l_b     = (const float*)d_in[10];
    const float* l2h_W     = (const float*)d_in[11];
    const float* l2h_b     = (const float*)d_in[12];
    const float* red_h_W   = (const float*)d_in[13];
    const float* red_h_b   = (const float*)d_in[14];
    const float* red_c_W   = (const float*)d_in[15];
    const float* red_c_b   = (const float*)d_in[16];
    const float* W_h       = (const float*)d_in[17];
    const float* W_c       = (const float*)d_in[18];
    const float* dp_W      = (const float*)d_in[19];
    const float* dp_b      = (const float*)d_in[20];
    const float* v_p       = (const float*)d_in[21];
    const float* xc_W      = (const float*)d_in[22];
    const float* xc_b      = (const float*)d_in[23];
    const float* dec_Wih   = (const float*)d_in[24];
    const float* dec_Whh   = (const float*)d_in[25];
    const float* dec_b     = (const float*)d_in[26];
    const float* pg_W      = (const float*)d_in[27];
    const float* pg_b      = (const float*)d_in[28];
    const float* out1_W    = (const float*)d_in[29];
    const float* out1_b    = (const float*)d_in[30];
    const float* out2_W    = (const float*)d_in[31];
    const float* out2_b    = (const float*)d_in[32];
    const int*   enc_input = (const int*)d_in[33];
    const int*   dec_input = (const int*)d_in[34];
    const int*   dec_tgt   = (const int*)d_in[35];
    const int*   enc_ext   = (const int*)d_in[36];

    // ---- workspace layout (~140 MB)
    Arena ar{(char*)d_ws, 0};
    h16*  encWf16 = (h16*)ar.get((size_t)2048 * 768 * 2);
    h16*  encWb16 = (h16*)ar.get((size_t)2048 * 768 * 2);
    h16*  decW16  = (h16*)ar.get((size_t)2048 * 768 * 2);
    h16*  h2l16   = (h16*)ar.get((size_t)128 * kH2 * 2);
    h16*  l2h16   = (h16*)ar.get((size_t)kH2 * 128 * 2);
    h16*  redh16  = (h16*)ar.get((size_t)kH * kH2 * 2);
    h16*  redc16  = (h16*)ar.get((size_t)kH * kH2 * 2);
    h16*  Wh16    = (h16*)ar.get((size_t)kH2 * kH2 * 2);
    h16*  dp16    = (h16*)ar.get((size_t)kH2 * kH2 * 2);
    h16*  xc16    = (h16*)ar.get((size_t)kE * (kH2 + kE) * 2);
    h16*  out116  = (h16*)ar.get((size_t)kH * 3 * kH * 2);
    h16*  out216  = (h16*)ar.get((size_t)kV * kH * 2);
    h16*  enc_out16 = (h16*)ar.get((size_t)kB * kLE * kH2 * 2);
    float* z32    = (float*)ar.get((size_t)kB * kLE * 128 * 4);
    h16*  z16     = (h16*)ar.get((size_t)kB * kLE * 128 * 2);
    float* zenc32 = (float*)ar.get((size_t)kB * kLE * kH2 * 4);
    h16*  zenc16  = (h16*)ar.get((size_t)kB * kLE * kH2 * 2);
    float* encfea = (float*)ar.get((size_t)kB * kLE * kH2 * 4);
    h16*  hst_f   = (h16*)ar.get((size_t)kB * kH * 2);
    h16*  hst_b   = (h16*)ar.get((size_t)kB * kH * 2);
    float* cst_f  = (float*)ar.get((size_t)kB * kH * 4);
    float* cst_b  = (float*)ar.get((size_t)kB * kH * 4);
    h16*  dec_h16 = (h16*)ar.get((size_t)kB * kH * 2);
    float* dec_c32= (float*)ar.get((size_t)kB * kH * 4);
    h16*  xh      = (h16*)ar.get((size_t)kB * 768 * 2);
    h16*  xh2     = (h16*)ar.get((size_t)kB * 768 * 2);
    float* G      = (float*)ar.get((size_t)kB * 2048 * 4);
    float* G2     = (float*)ar.get((size_t)kB * 2048 * 4);
    h16*  hTcat   = (h16*)ar.get((size_t)kB * kH2 * 2);
    h16*  cTcat   = (h16*)ar.get((size_t)kB * kH2 * 2);
    float* sh0    = (float*)ar.get((size_t)kB * kH * 4);
    float* shat32 = (float*)ar.get((size_t)kB * kH2 * 4);
    h16*  shat16  = (h16*)ar.get((size_t)kB * kH2 * 2);
    float* decfea = (float*)ar.get((size_t)kB * kH2 * 4);
    float* scores = (float*)ar.get((size_t)kB * kLE * 4);
    float* attn   = (float*)ar.get((size_t)kB * kLE * 4);
    float* cov    = (float*)ar.get((size_t)kB * kLE * 4);
    float* covls  = (float*)ar.get((size_t)kB * kLE * 4);
    float* covterm= (float*)ar.get((size_t)kB * 4);
    float* pgen   = (float*)ar.get((size_t)kB * 4);
    float* acc    = (float*)ar.get((size_t)kB * 4);
    float* ctx    = (float*)ar.get((size_t)kB * kH2 * 4);
    h16*  xcin    = (h16*)ar.get((size_t)kB * (kH2 + kE) * 2);
    float* x32    = (float*)ar.get((size_t)kB * kE * 4);
    h16*  x16     = (h16*)ar.get((size_t)kB * kE * 2);
    h16*  out1in  = (h16*)ar.get((size_t)kB * 3 * kH * 2);
    float* out32  = (float*)ar.get((size_t)kB * kH * 4);
    h16*  o16     = (h16*)ar.get((size_t)kB * kH * 2);
    float* logits = (float*)ar.get((size_t)kB * kV * 4);
    float* fin    = (float*)ar.get((size_t)kB * kVX * 4);

    auto gemm = [&](const h16* A, int lda, const h16* Wt, int ldw, const float* bias,
                    float* C, h16* Ch, int ldc, int M, int N, int K, int act) {
        int tiles = (M / 16) * (N / 64);                 // 16x64 tile per wave
        int nb = (tiles + 7) / 8;                        // 8 waves per 256-thread block
        gemm64_k<<<nb, 256, 0, stream>>>(A, lda, Wt, ldw, bias, C, Ch, ldc, N, K, act, tiles);
    };
    auto cvt = [&](const float* s, h16* d, long n) {
        cvt_f16_k<<<blks(n), 256, 0, stream>>>(s, d, n);
    };

    // ---- weight conversion (f32 -> f16), once per launch
    catw_k<<<blks((long)2048 * 768), 256, 0, stream>>>(enc_Wih_f, enc_Whh_f, encWf16, 2048, kE, kH);
    catw_k<<<blks((long)2048 * 768), 256, 0, stream>>>(enc_Wih_b, enc_Whh_b, encWb16, 2048, kE, kH);
    catw_k<<<blks((long)2048 * 768), 256, 0, stream>>>(dec_Wih, dec_Whh, decW16, 2048, kE, kH);
    cvt(h2l_W, h2l16, (long)128 * kH2);
    cvt(l2h_W, l2h16, (long)kH2 * 128);
    cvt(red_h_W, redh16, (long)kH * kH2);
    cvt(red_c_W, redc16, (long)kH * kH2);
    cvt(W_h, Wh16, (long)kH2 * kH2);
    cvt(dp_W, dp16, (long)kH2 * kH2);
    cvt(xc_W, xc16, (long)kE * (kH2 + kE));
    cvt(out1_W, out116, (long)kH * 3 * kH);
    cvt(out2_W, out216, (long)kV * kH);

    // ---- state init
    zero_f16_k<<<blks(kB * kH), 256, 0, stream>>>(hst_f, kB * kH);
    zero_f16_k<<<blks(kB * kH), 256, 0, stream>>>(hst_b, kB * kH);
    zero_f32_k<<<blks(kB * kH), 256, 0, stream>>>(cst_f, kB * kH);
    zero_f32_k<<<blks(kB * kH), 256, 0, stream>>>(cst_b, kB * kH);
    zero_f32_k<<<blks(kB * kLE), 256, 0, stream>>>(cov, kB * kLE);
    zero_f32_k<<<blks(kB * kLE), 256, 0, stream>>>(covls, kB * kLE);
    zero_f32_k<<<1, 256, 0, stream>>>(acc, kB);
    zero_f32_k<<<blks(kB * kH2), 256, 0, stream>>>(ctx, kB * kH2);

    // ---- BiLSTM encoder (fwd + bwd per step, fused [x|h]*[Wih|Whh]^T GEMM)
    for (int t = 0; t < kLE; ++t) {
        enc_x_k<<<48, 256, 0, stream>>>(emb, enc_input, t, 0, hst_f, xh);
        gemm(xh, 768, encWf16, 768, nullptr, G, nullptr, 2048, kB, 2048, 768, 0);
        cell_k<<<32, 256, 0, stream>>>(G, enc_b_f, cst_f, hst_f,
                                       enc_out16 + (size_t)t * kH2, kLE * kH2,
                                       nullptr, nullptr);
        enc_x_k<<<48, 256, 0, stream>>>(emb, enc_input, t, 1, hst_b, xh2);
        gemm(xh2, 768, encWb16, 768, nullptr, G2, nullptr, 2048, kB, 2048, 768, 0);
        cell_k<<<32, 256, 0, stream>>>(G2, enc_b_b, cst_b, hst_b,
                                       enc_out16 + (size_t)(kLE - 1 - t) * kH2 + kH, kLE * kH2,
                                       nullptr, nullptr);
    }

    // ---- latent chain over all 6400 positions (big parallel WMMA GEMMs)
    gemm(enc_out16, kH2, h2l16, kH2, h2l_b, z32, z16, 128, kB * kLE, 128, kH2, 0);
    gemm(z16, 128, l2h16, 128, l2h_b, zenc32, zenc16, kH2, kB * kLE, kH2, 128, 0);
    gemm(zenc16, kH2, Wh16, kH2, nullptr, encfea, nullptr, kH2, kB * kLE, kH2, kH2, 0);

    // ---- reduce_state -> decoder initial (s_h, s_c)
    pack2h_k<<<blks(kB * kH2), 256, 0, stream>>>(hst_f, kH, hst_b, kH, hTcat, kB);
    pack2f_k<<<blks(kB * kH2), 256, 0, stream>>>(cst_f, kH, kH, cst_b, kH, kH, cTcat, kB);
    gemm(hTcat, kH2, redh16, kH2, red_h_b, sh0, dec_h16, kH, kB, kH, kH2, 1);
    gemm(cTcat, kH2, redc16, kH2, red_c_b, dec_c32, nullptr, kH, kB, kH, kH2, 1);
    pack2f_k<<<blks(kB * kH2), 256, 0, stream>>>(sh0, kH, kH, dec_c32, kH, kH, shat16, kB);

    // ---- step-0 extra attention (c_t discarded, coverage kept; covloss untouched)
    gemm(shat16, kH2, dp16, kH2, dp_b, decfea, nullptr, kH2, kB, kH2, kH2, 0);
    attn_score_k<<<800, 256, 0, stream>>>(encfea, decfea, cov, W_c, v_p, scores);
    attn_softmax_k<<<kB, 512, 0, stream>>>(scores, enc_mask, attn, cov, covls, covterm, 0);

    // ---- decoder (64 sequential steps)
    for (int t = 0; t < kLD; ++t) {
        dec_x_k<<<80, 256, 0, stream>>>(ctx, emb, dec_input, t, xcin);        // [c_t1 | y_emb]
        gemm(xcin, kH2 + kE, xc16, kH2 + kE, xc_b, x32, x16, kE, kB, kE, kH2 + kE, 0);
        pack2h_k<<<48, 256, 0, stream>>>(x16, kE, dec_h16, kH, xh, kB);       // [x | h]
        gemm(xh, kE + kH, decW16, kE + kH, nullptr, G, nullptr, 2048, kB, 2048, kE + kH, 0);
        cell_k<<<32, 256, 0, stream>>>(G, dec_b, dec_c32, dec_h16, nullptr, 0, shat32, shat16);
        gemm(shat16, kH2, dp16, kH2, dp_b, decfea, nullptr, kH2, kB, kH2, kH2, 0);
        attn_score_k<<<800, 256, 0, stream>>>(encfea, decfea, cov, W_c, v_p, scores);
        attn_softmax_k<<<kB, 512, 0, stream>>>(scores, enc_mask, attn, cov, covls, covterm, 1);
        context_k<<<kB, 256, 0, stream>>>(attn, zenc32, ctx);                 // new c_t
        pack2f_k<<<96, 256, 0, stream>>>(shat32, kH2, kH, ctx, kH2, kH2, out1in, kB);
        gemm(out1in, 3 * kH, out116, 3 * kH, out1_b, out32, o16, kH, kB, kH, 3 * kH, 0);
        gemm(o16, kH, out216, kH, out2_b, logits, nullptr, kV, kB, kV, kH, 0);
        pgen_k<<<kB, 256, 0, stream>>>(ctx, shat32, x32, pg_W, pg_b, pgen);
        vocab_dist_k<<<kB, 512, 0, stream>>>(logits, pgen, fin);
        scatter_k<<<25, 256, 0, stream>>>(attn, pgen, enc_ext, fin);
        nll_k<<<1, 32, 0, stream>>>(fin, dec_tgt, dec_mask, covterm, t, acc);
    }

    finalize_k<<<1, 32, 0, stream>>>(acc, dec_mask, (float*)d_out);
}
